// MultiHeadSelfAttention_11759620456627
// MI455X (gfx1250) — matrix operand order repaired and checked by reading, not compiled, not run on hardware
//
#include <hip/hip_runtime.h>

// ---------------------------------------------------------------------------
// MHA forward for gfx1250 (MI455X), bf16 WMMA 16x16x32 everywhere.
// Shapes: B=2, L=2048, D=1024, H=16, d_k=64.
// ---------------------------------------------------------------------------

typedef __bf16 bf16;
typedef __attribute__((ext_vector_type(16))) __bf16 v16bf;
typedef __attribute__((ext_vector_type(8)))  __bf16 v8bfv;
typedef __attribute__((ext_vector_type(8)))  float  v8f;

#define DEV __device__ __forceinline__

static constexpr int Bsz = 2;
static constexpr int L   = 2048;
static constexpr int D   = 1024;
static constexpr int H   = 16;
static constexpr int DK  = 64;
static constexpr int E3  = 3 * D;         // 3072

DEV v8f wmma_bf16(v16bf a, v16bf b, v8f c) {
  return __builtin_amdgcn_wmma_f32_16x16x32_bf16(false, a, false, b, (short)0, c,
                                                 false, false);
}

// A fragment (16x32, M x K), row-major source with row pitch `pitch`.
// Lane L (m=L&15, hk=L>>4) holds K = {hk*8..hk*8+7, 16+hk*8..16+hk*8+7}.
DEV v16bf load_frag_a(const bf16* tile, int pitch, int lane) {
  const int m = lane & 15, hk = lane >> 4;
  const bf16* p = tile + (size_t)m * pitch + hk * 8;
  v8bfv c0 = *(const v8bfv*)(p);
  v8bfv c1 = *(const v8bfv*)(p + 16);
  v16bf r;
#pragma unroll
  for (int i = 0; i < 8; ++i) { r[i] = c0[i]; r[i + 8] = c1[i]; }
  return r;
}

// B fragment (32x16, K x N) sourced from Bt (N x K row-major, pitch `pitch`).
// Lane L (n=L&15, hk=L>>4) holds K = {hk*8..hk*8+7, 16+hk*8..16+hk*8+7}.
DEV v16bf load_frag_b(const bf16* tileT, int pitch, int lane) {
  const int n = lane & 15, hk = lane >> 4;
  const bf16* p = tileT + (size_t)n * pitch + hk * 8;
  v8bfv c0 = *(const v8bfv*)(p);
  v8bfv c1 = *(const v8bfv*)(p + 16);
  v16bf r;
#pragma unroll
  for (int i = 0; i < 8; ++i) { r[i] = c0[i]; r[i + 8] = c1[i]; }
  return r;
}

// 16-lane (half-wave) row reductions (C/D rows live inside one 16-lane group).
DEV float rowmax16(float v) {
  v = fmaxf(v, __shfl_xor(v, 1, 32));
  v = fmaxf(v, __shfl_xor(v, 2, 32));
  v = fmaxf(v, __shfl_xor(v, 4, 32));
  v = fmaxf(v, __shfl_xor(v, 8, 32));
  return v;
}
DEV float rowsum16(float v) {
  v += __shfl_xor(v, 1, 32);
  v += __shfl_xor(v, 2, 32);
  v += __shfl_xor(v, 4, 32);
  v += __shfl_xor(v, 8, 32);
  return v;
}

// 64x32 output macro-tile: acc[4][2], B-fragments live across mi, A transient.
// Peak VGPR ~= 64 (acc) + 16 (B) + 8 (A) + addressing -> no spills.
DEV void gemm_64x32(const bf16* A, const bf16* Bt, int K, int lda, int ldb,
                    int lane, v8f acc[4][2]) {
  for (int k = 0; k < K; k += 32) {
    const v16bf b0 = load_frag_b(Bt + k, ldb, lane);
    const v16bf b1 = load_frag_b(Bt + (size_t)16 * ldb + k, ldb, lane);
#pragma unroll
    for (int mi = 0; mi < 4; ++mi) {
      const v16bf a = load_frag_a(A + (size_t)(mi * 16) * lda + k, lda, lane);
      acc[mi][0] = wmma_bf16(a, b0, acc[mi][0]);
      acc[mi][1] = wmma_bf16(a, b1, acc[mi][1]);
    }
  }
}

// ---------------------------------------------------------------------------
// Kernel 0: weight transpose. dst[c*rows + r] = src[r*cols + c]
// ---------------------------------------------------------------------------
__global__ void k_transpose(const bf16* __restrict__ src, bf16* __restrict__ dst,
                            int rows, int cols) {
  int idx = blockIdx.x * 256 + threadIdx.x;
  if (idx < rows * cols) {
    int r = idx / cols, c = idx % cols;
    dst[(size_t)c * rows + r] = src[idx];
  }
}

// ---------------------------------------------------------------------------
// Kernel 1: qkv = x @ W_qkv. One wave per 64x32 output tile (8 wmma / K-step).
// Scatter into Q,K (B,H,L,64) and Vt (B,H,64,L). 32-aligned column tiles never
// straddle the q/k/v or head boundaries, so c,h are wave-uniform.
// ---------------------------------------------------------------------------
__global__ void k_qkv_gemm(const bf16* __restrict__ x, const bf16* __restrict__ WqkvT,
                           bf16* __restrict__ Qw, bf16* __restrict__ Kw,
                           bf16* __restrict__ Vtw) {
  const int lane = threadIdx.x & 31;
  const int wave = threadIdx.x >> 5;
  const int tile = blockIdx.x * 8 + wave;     // 64 * 96 = 6144 tiles
  const int tn = tile % (E3 / 32);
  const int tm = tile / (E3 / 32);

  v8f acc[4][2] = {};
  gemm_64x32(x + (size_t)tm * 64 * D, WqkvT + (size_t)tn * 32 * D, D, D, D,
             lane, acc);

  const int n = lane & 15, hk = lane >> 4;
  const int e0 = tn * 32;                     // column base (32-aligned)
  const int c  = e0 >> 10;                    // 0=Q 1=K 2=V (uniform)
  const int h  = (e0 & 1023) >> 6;            // head (uniform)
  const int d0 = e0 & 63;                     // 0 or 32 (uniform)
#pragma unroll
  for (int mi = 0; mi < 4; ++mi)
#pragma unroll
    for (int ni = 0; ni < 2; ++ni)
#pragma unroll
      for (int r = 0; r < 8; ++r) {
        const int t  = tm * 64 + mi * 16 + r + 8 * hk;  // row = b*L + l
        const int b_ = t >> 11;
        const int li = t & (L - 1);
        const int d  = d0 + ni * 16 + n;
        const size_t head = (size_t)(b_ * H + h);
        const bf16 v = (bf16)acc[mi][ni][r];
        if (c == 0)      Qw [(head * L + li) * DK + d] = v;
        else if (c == 1) Kw [(head * L + li) * DK + d] = v;
        else             Vtw[(head * DK + d) * L + li] = v;
      }
}

// ---------------------------------------------------------------------------
// Kernel 2: causal flash attention. 4 waves/block; one wave owns 32 queries
// (two 16-row M-subtiles) -> K/V fragments are reused, 16 wmma per 32-key tile.
// K-fragments kept transient to stay under the VGPR budget.
// ---------------------------------------------------------------------------
__global__ void k_attn(const bf16* __restrict__ Q, const bf16* __restrict__ K,
                       const bf16* __restrict__ Vt, bf16* __restrict__ ctx) {
  __shared__ __align__(16) bf16 pbuf[4][2 * 16 * 32];   // per-wave P (32x32 bf16)

  const int lane = threadIdx.x & 31;
  const int wave = threadIdx.x >> 5;
  const int bh   = blockIdx.y;                          // b*H + h
  const int q_base = (blockIdx.x * 4 + wave) * 32;
  const int h = bh & (H - 1);
  const int b = bh >> 4;

  const bf16* Qh = Q  + (size_t)bh * L * DK;
  const bf16* Kh = K  + (size_t)bh * L * DK;
  const bf16* Vh = Vt + (size_t)bh * DK * L;

  // Q fragments: [mi][k-half], reused across all key tiles.
  v16bf qa[2][2];
#pragma unroll
  for (int mi = 0; mi < 2; ++mi) {
    qa[mi][0] = load_frag_a(Qh + (size_t)(q_base + mi * 16) * DK +  0, DK, lane);
    qa[mi][1] = load_frag_a(Qh + (size_t)(q_base + mi * 16) * DK + 32, DK, lane);
  }

  const int n  = lane & 15;
  const int hk = lane >> 4;

  float m2[2][8], lsum[2][8];
  v8f acc[2][4] = {};
#pragma unroll
  for (int mi = 0; mi < 2; ++mi)
#pragma unroll
    for (int r = 0; r < 8; ++r) { m2[mi][r] = -1e30f; lsum[mi][r] = 0.0f; }

  const float sl2e = 0.125f * 1.4426950408889634f;  // rsqrt(64) * log2(e)
  const int kend = q_base + 32;                     // causal bound (exclusive)

  for (int kb = 0; kb < kend; kb += 32) {
    // ---- S = Q K^T; K fragments transient per 16-key sub-tile ----
    v8f s[2][2];
#pragma unroll
    for (int nt = 0; nt < 2; ++nt) {
      const v16bf kf0 = load_frag_b(Kh + (size_t)(kb + nt * 16) * DK +  0, DK, lane);
      const v16bf kf1 = load_frag_b(Kh + (size_t)(kb + nt * 16) * DK + 32, DK, lane);
#pragma unroll
      for (int mi = 0; mi < 2; ++mi) {
        v8f t = {};
        t = wmma_bf16(qa[mi][0], kf0, t);
        t = wmma_bf16(qa[mi][1], kf1, t);
        s[mi][nt] = t;
      }
    }
    if (kb + 32 < kend) {
      __builtin_prefetch(Kh + (size_t)(kb + 32) * DK, 0, 1);
      __builtin_prefetch(Vh + (size_t)kb + 32, 0, 1);
    }

    // ---- online softmax (fp32, exp2 domain), causal mask, P -> LDS ----
#pragma unroll
    for (int mi = 0; mi < 2; ++mi) {
      bf16* pdst = &pbuf[wave][mi * 16 * 32];
#pragma unroll
      for (int r = 0; r < 8; ++r) {
        const int q  = q_base + mi * 16 + r + 8 * hk;
        const int k0 = kb + n;
        const int k1 = kb + 16 + n;
        const float a0 = (k0 <= q) ? s[mi][0][r] * sl2e : -1e30f;
        const float a1 = (k1 <= q) ? s[mi][1][r] * sl2e : -1e30f;

        const float tmax = rowmax16(fmaxf(a0, a1));
        const float mnew = fmaxf(m2[mi][r], tmax);
        const float resc = __builtin_amdgcn_exp2f(m2[mi][r] - mnew);
        m2[mi][r] = mnew;

        const float p0 = __builtin_amdgcn_exp2f(a0 - mnew);
        const float p1 = __builtin_amdgcn_exp2f(a1 - mnew);
        lsum[mi][r] = lsum[mi][r] * resc + rowsum16(p0 + p1);
#pragma unroll
        for (int c = 0; c < 4; ++c) acc[mi][c][r] *= resc;

        const int m = r + 8 * hk;
        pdst[m * 32 + n]      = (bf16)p0;
        pdst[m * 32 + 16 + n] = (bf16)p1;
      }
    }

    // DS ops of one wave process in order; explicit fence for paranoia.
    asm volatile("s_wait_dscnt 0" ::: "memory");

    // ---- ctx += P V : V fragments reused across both M-subtiles ----
    v16bf pa[2];
#pragma unroll
    for (int mi = 0; mi < 2; ++mi)
      pa[mi] = load_frag_a(&pbuf[wave][mi * 16 * 32], 32, lane);
#pragma unroll
    for (int c = 0; c < 4; ++c) {
      const v16bf vb = load_frag_b(Vh + (size_t)(c * 16) * L + kb, L, lane);
#pragma unroll
      for (int mi = 0; mi < 2; ++mi)
        acc[mi][c] = wmma_bf16(pa[mi], vb, acc[mi][c]);
    }
  }

  // ---- normalize and store ctx tile (rows = queries, cols = h*64 + d) ----
#pragma unroll
  for (int mi = 0; mi < 2; ++mi)
#pragma unroll
    for (int r = 0; r < 8; ++r) {
      const float inv = 1.0f / lsum[mi][r];
      const int rowg = b * L + q_base + mi * 16 + r + 8 * hk;
#pragma unroll
      for (int c = 0; c < 4; ++c) {
        const int col = h * DK + c * 16 + n;
        ctx[(size_t)rowg * D + col] = (bf16)(acc[mi][c][r] * inv);
      }
    }
}

// ---------------------------------------------------------------------------
// Kernel 3: out = ctx @ W_out. One wave per 64x32 output tile.
// ---------------------------------------------------------------------------
__global__ void k_out_gemm(const bf16* __restrict__ ctx, const bf16* __restrict__ WoutT,
                           bf16* __restrict__ out) {
  const int lane = threadIdx.x & 31;
  const int wave = threadIdx.x >> 5;
  const int tile = blockIdx.x * 8 + wave;     // 64 * 32 = 2048 tiles
  const int tn = tile % (D / 32);
  const int tm = tile / (D / 32);

  v8f acc[4][2] = {};
  gemm_64x32(ctx + (size_t)tm * 64 * D, WoutT + (size_t)tn * 32 * D, D, D, D,
             lane, acc);

  const int n = lane & 15, hk = lane >> 4;
#pragma unroll
  for (int mi = 0; mi < 4; ++mi)
#pragma unroll
    for (int ni = 0; ni < 2; ++ni)
#pragma unroll
      for (int r = 0; r < 8; ++r) {
        const int row = tm * 64 + mi * 16 + r + 8 * hk;
        out[(size_t)row * D + tn * 32 + ni * 16 + n] = (bf16)acc[mi][ni][r];
      }
}

// ---------------------------------------------------------------------------
// Host launcher. Inputs: x (bf16), mask (ignored; causal), W_qkv, W_out (bf16).
// Workspace layout (bytes):
//   [0      , 6 MB)   WqkvT  (3072 x 1024)
//   [6 MB   , 8 MB)   WoutT  (1024 x 1024)
//   [8 MB   , 16 MB)  Q      (B,H,L,64)
//   [16 MB  , 24 MB)  K      (B,H,L,64)
//   [24 MB  , 32 MB)  Vt     (B,H,64,L)
//   [32 MB  , 40 MB)  ctx    (B*L, D)
// ---------------------------------------------------------------------------
extern "C" void kernel_launch(void* const* d_in, const int* in_sizes, int n_in,
                              void* d_out, int out_size, void* d_ws, size_t ws_size,
                              hipStream_t stream) {
  (void)in_sizes; (void)n_in; (void)out_size; (void)ws_size;
  const bf16* x    = (const bf16*)d_in[0];
  const bf16* Wqkv = (const bf16*)d_in[2];
  const bf16* Wout = (const bf16*)d_in[3];
  bf16* out = (bf16*)d_out;

  char* ws = (char*)d_ws;
  bf16* WqkvT = (bf16*)(ws);
  bf16* WoutT = (bf16*)(ws + (size_t)6  * 1024 * 1024);
  bf16* Qw    = (bf16*)(ws + (size_t)8  * 1024 * 1024);
  bf16* Kw    = (bf16*)(ws + (size_t)16 * 1024 * 1024);
  bf16* Vtw   = (bf16*)(ws + (size_t)24 * 1024 * 1024);
  bf16* ctx   = (bf16*)(ws + (size_t)32 * 1024 * 1024);

  k_transpose<<<dim3((D * E3) / 256), dim3(256), 0, stream>>>(Wqkv, WqkvT, D, E3);
  k_transpose<<<dim3((D * D) / 256),  dim3(256), 0, stream>>>(Wout, WoutT, D, D);

  // 4096/64 * 3072/32 = 6144 tiles, 8 waves per 256-thread block
  k_qkv_gemm<<<dim3(6144 / 8), dim3(256), 0, stream>>>(x, WqkvT, Qw, Kw, Vtw);

  // grid: (L/128 query blocks, B*H head-batches), 4 waves * 32 queries each
  k_attn<<<dim3(L / 128, Bsz * H), dim3(128), 0, stream>>>(Qw, Kw, Vtw, ctx);

  // 4096/64 * 1024/32 = 2048 tiles
  k_out_gemm<<<dim3(2048 / 8), dim3(256), 0, stream>>>(ctx, WoutT, out);
}
